// SGCN_52518860095865
// MI455X (gfx1250) — matrix operand order repaired and checked by reading, not compiled, not run on hardware
//
#include <hip/hip_runtime.h>
#include <hip/hip_bf16.h>
#include <math.h>

// ---------------------------------------------------------------------------
// CDNA5 (gfx1250): 3-layer BiLSTM + cosine-GCN + pool/FC head.
// All GEMMs: V_WMMA_F32_16X16X32_BF16, bf16 operands pre-staged in workspace,
// f32 accumulate. GEMM inner loop = global_load_b128 -> v_wmma (no cvt VALU).
// ---------------------------------------------------------------------------

typedef __attribute__((ext_vector_type(16))) __bf16 v16bf;
typedef __attribute__((ext_vector_type(8)))  __bf16 v8bf;
typedef __attribute__((ext_vector_type(8)))  float  v8f;

__device__ __forceinline__ __bf16 f2bf(float f) {
  union { float f; unsigned u; } in; in.f = f;
  unsigned r = in.u + 0x7FFFu + ((in.u >> 16) & 1u);   // round-to-nearest-even
  union { unsigned short s; __bf16 b; } out; out.s = (unsigned short)(r >> 16);
  return out.b;
}

__device__ __forceinline__ float sigf(float x) { return 1.f / (1.f + __expf(-x)); }

// ---------------------------------------------------------------------------
// Batched GEMM:  C[m,n] = sum_k A[m,k]*W[n,k] (+bias1[n]+bias2[n])
//   A (bf16) element (m,k) at A + z*sA + m*lda + k
//   W (bf16) element (n,k) at W + z*sW + n*ldw + k
//   C (f32)  element (m,n) at C + z*sC + m*ldc + n
// M%32==0, N%128==0, K%32==0; lda/ldw multiples of 16 elems; 64B-aligned bases.
// Block = 256 threads = 8 waves; wave = two adjacent 16x16 N-tiles sharing the
// A fragment (2 WMMAs per A load); block tile = 32x128.
// ---------------------------------------------------------------------------
__global__ __launch_bounds__(256) void gemm_wmma(
    const __bf16* __restrict__ A, long long sA, int lda,
    const __bf16* __restrict__ W, long long sW, int ldw,
    const float* __restrict__ b1, const float* __restrict__ b2,
    float* __restrict__ C, long long sC, int ldc, int K)
{
  int z = blockIdx.z;
  A += (long long)z * sA; W += (long long)z * sW; C += (long long)z * sC;
  int wave = threadIdx.x >> 5;
  int lane = threadIdx.x & 31;
  int half = lane >> 4;          // 16-lane group
  int r    = lane & 15;
  int tm = blockIdx.y * 32  + (wave >> 2) * 16;
  int tn = blockIdx.x * 128 + (wave & 3) * 32;

  const __bf16* arow  = A + (long long)(tm + r) * lda + 16 * half;   // A: row m
  const __bf16* wrow0 = W + (long long)(tn + r) * ldw + 16 * half;   // B: col n
  const __bf16* wrow1 = wrow0 + 16 * ldw;                            // col n+16

  float bb0 = 0.f, bb1 = 0.f;
  if (b1) { bb0 += b1[tn + r]; bb1 += b1[tn + 16 + r]; }
  if (b2) { bb0 += b2[tn + r]; bb1 += b2[tn + 16 + r]; }
  v8f acc0 = {bb0, bb0, bb0, bb0, bb0, bb0, bb0, bb0};
  v8f acc1 = {bb1, bb1, bb1, bb1, bb1, bb1, bb1, bb1};

  for (int k0 = 0; k0 < K; k0 += 32) {
    __builtin_prefetch(arow + k0 + 128, 0, 1);      // global_prefetch_b8
    __builtin_prefetch(wrow0 + k0 + 128, 0, 1);
    v8bf alo = *(const v8bf*)(arow + k0);           // A layout: k = e + 16*half
    v8bf ahi = *(const v8bf*)(arow + k0 + 8);
    v16bf af = __builtin_shufflevector(alo, ahi, 0, 1, 2, 3, 4, 5, 6, 7,
                                       8, 9, 10, 11, 12, 13, 14, 15);
    v16bf wf0 = *(const v16bf*)(wrow0 + k0);        // B layout: k = e + 16*half
    v16bf wf1 = *(const v16bf*)(wrow1 + k0);
    acc0 = __builtin_amdgcn_wmma_f32_16x16x32_bf16(false, af, false, wf0,
                                                   (short)0, acc0, false, false);
    acc1 = __builtin_amdgcn_wmma_f32_16x16x32_bf16(false, af, false, wf1,
                                                   (short)0, acc1, false, false);
  }
#pragma unroll
  for (int q = 0; q < 8; ++q) {                     // D: m = q + 8*half, n = lane&15
    float* crow = C + (long long)(tm + q + 8 * half) * ldc + tn + r;
    crow[0]  = acc0[q];
    crow[16] = acc1[q];
  }
}

// ---------------------------------------------------------------------------
// Persistent BiLSTM layer. One block per (input, direction) sequence.
// xpre: [2 inp][2 dir][S][64][512], whh: [2 dir][512][128] (f32),
// hout (f32) + houtb (bf16): [2 inp][S][64][256], dir d -> cols d*128..+127.
// LDS: W_hh bf16 (128KB) + h bf16 (16KB) + gate f32 (128KB) = 272KB/WGP.
// Cell state c: 16 f32 VGPRs/thread (512 thr * 16 = 64*128).
// ---------------------------------------------------------------------------
__global__ __launch_bounds__(512) void lstm_layer(
    const float* __restrict__ xpre, const float* __restrict__ whh,
    float* __restrict__ hout, __bf16* __restrict__ houtb, int S)
{
  int inp = blockIdx.x >> 1, dir = blockIdx.x & 1;
  xpre  += (long long)(inp * 2 + dir) * S * 64 * 512;
  whh   += (long long)dir * 512 * 128;
  hout  += (long long)inp * S * 64 * 256 + dir * 128;
  houtb += (long long)inp * S * 64 * 256 + dir * 128;

  __shared__ __attribute__((aligned(64))) __bf16 s_whh[512 * 128];
  __shared__ __attribute__((aligned(64))) __bf16 s_h[64 * 128];
  __shared__ __attribute__((aligned(64))) float  s_g[64 * 512];

  int tid = threadIdx.x;
  for (int i = tid; i < 512 * 128; i += 512) s_whh[i] = f2bf(whh[i]);
  for (int i = tid; i < 64 * 128;  i += 512) s_h[i]   = f2bf(0.f);
  float c[16];
#pragma unroll
  for (int q = 0; q < 16; ++q) c[q] = 0.f;
  __syncthreads();

  int wave = tid >> 5, lane = tid & 31, half = lane >> 4, r = lane & 15;

  for (int step = 0; step < S; ++step) {
    int t = dir ? (S - 1 - step) : step;
    const float* xp = xpre + (long long)t * 64 * 512;

    // ---- GEMM: s_g = h @ W_hh^T (64x512x128); 16 waves x 8 tiles ----
    for (int T = wave * 8; T < wave * 8 + 8; ++T) {
      int tm = (T >> 5) << 4;      // 0..48
      int tn = (T & 31) << 4;      // 0..496
      const __bf16* hrow = s_h   + (tm + r) * 128 + 16 * half;
      const __bf16* wrw  = s_whh + (tn + r) * 128 + 16 * half;
      v8f acc = {};
#pragma unroll
      for (int k0 = 0; k0 < 128; k0 += 32) {
        v8bf lo = *(const v8bf*)(hrow + k0);        // ds_load_b128
        v8bf hi = *(const v8bf*)(hrow + k0 + 8);
        v16bf af = __builtin_shufflevector(lo, hi, 0, 1, 2, 3, 4, 5, 6, 7,
                                           8, 9, 10, 11, 12, 13, 14, 15);
        v16bf wf = *(const v16bf*)(wrw + k0);
        acc = __builtin_amdgcn_wmma_f32_16x16x32_bf16(false, af, false, wf,
                                                      (short)0, acc, false, false);
      }
#pragma unroll
      for (int q = 0; q < 8; ++q)
        s_g[(tm + q + 8 * half) * 512 + tn + r] = acc[q];
    }
    __syncthreads();

    // ---- gates: torch order i,f,g,o ----
#pragma unroll
    for (int q = 0; q < 16; ++q) {
      int idx = tid * 16 + q;          // (m, j) over 64*128
      int m = idx >> 7, j = idx & 127;
      const float* xr = xp + m * 512;
      float gi = s_g[m * 512 +       j] + xr[      j];
      float gf = s_g[m * 512 + 128 + j] + xr[128 + j];
      float gg = s_g[m * 512 + 256 + j] + xr[256 + j];
      float go = s_g[m * 512 + 384 + j] + xr[384 + j];
      gi = sigf(gi); gf = sigf(gf); gg = tanhf(gg); go = sigf(go);
      c[q] = gf * c[q] + gi * gg;
      float h = go * tanhf(c[q]);
      __bf16 hb = f2bf(h);
      hout [(long long)t * 64 * 256 + m * 256 + j] = h;
      houtb[(long long)t * 64 * 256 + m * 256 + j] = hb;
      s_h[m * 128 + j] = hb;
    }
    __syncthreads();
  }
}

// ---------------------------------------------------------------------------
// Elementwise / staging kernels
// ---------------------------------------------------------------------------

// [B,S,300] f32 -> [S,B,320] bf16, zero-padded K tail.
__global__ void transpose_pad(const float* __restrict__ x, __bf16* __restrict__ xT,
                              int S, int B) {
  long long n = (long long)S * B * 320;
  for (long long i = (long long)blockIdx.x * blockDim.x + threadIdx.x; i < n;
       i += (long long)gridDim.x * blockDim.x) {
    int k = (int)(i % 320);
    long long sb = i / 320;
    int b = (int)(sb % B), s = (int)(sb / B);
    xT[i] = (k < 300) ? f2bf(x[((long long)b * S + s) * 300 + k]) : f2bf(0.f);
  }
}

// w_ih0 [2,512,300] f32 -> [2,512,320] bf16, zero-padded.
__global__ void pad_w0(const float* __restrict__ w, __bf16* __restrict__ wp) {
  long long n = 2ll * 512 * 320;
  for (long long i = (long long)blockIdx.x * blockDim.x + threadIdx.x; i < n;
       i += (long long)gridDim.x * blockDim.x) {
    int k = (int)(i % 320);
    long long dn = i / 320;
    wp[i] = (k < 300) ? f2bf(w[dn * 300 + k]) : f2bf(0.f);
  }
}

// generic f32 -> bf16 copy
__global__ void convert_bf(const float* __restrict__ s, __bf16* __restrict__ d,
                           long long n) {
  for (long long i = (long long)blockIdx.x * blockDim.x + threadIdx.x; i < n;
       i += (long long)gridDim.x * blockDim.x) d[i] = f2bf(s[i]);
}

// gcn_w [256,128] -> gcn_wT [128,256] bf16
__global__ void transpose_gcnw(const float* __restrict__ w, __bf16* __restrict__ wT) {
  int i = blockIdx.x * blockDim.x + threadIdx.x;      // over 128*256
  if (i >= 128 * 256) return;
  int d = i & 255, h = i >> 8;
  wT[i] = f2bf(w[d * 128 + h]);
}

__global__ void row_invnorm(const float* __restrict__ o, float* __restrict__ inv) {
  int gid = blockIdx.x * blockDim.x + threadIdx.x;    // over S*B = 16384 rows [s][b]
  if (gid >= 16384) return;
  int s = gid >> 6, b = gid & 63;
  const float* rp = o + (long long)gid * 256;
  float sum = 0.f;
  for (int k = 0; k < 256; ++k) { float v = rp[k]; sum += v * v; }
  inv[b * 256 + s] = 1.f / fmaxf(sqrtf(sum), 1e-12f);
}

// normalize M in place (f32) and emit bf16 copy.
__global__ void scale_M(float* __restrict__ M, __bf16* __restrict__ M16,
                        const float* __restrict__ inv1, const float* __restrict__ inv2) {
  long long n = 64ll * 256 * 256;
  for (long long i = (long long)blockIdx.x * blockDim.x + threadIdx.x; i < n;
       i += (long long)gridDim.x * blockDim.x) {
    int b = (int)(i >> 16), s = (int)((i >> 8) & 255), t = (int)(i & 255);
    float v = M[i] * inv1[b * 256 + s] * inv2[b * 256 + t];
    M[i] = v;
    M16[i] = f2bf(v);
  }
}

// per-batch transpose of normalized M -> bf16
__global__ void transpose_M(const float* __restrict__ M, __bf16* __restrict__ MT) {
  long long n = 64ll * 256 * 256;
  for (long long i = (long long)blockIdx.x * blockDim.x + threadIdx.x; i < n;
       i += (long long)gridDim.x * blockDim.x) {
    int b = (int)(i >> 16), t = (int)((i >> 8) & 255), s = (int)(i & 255);
    MT[i] = f2bf(M[(long long)b * 65536 + s * 256 + t]);
  }
}

__global__ void degree_d(const float* __restrict__ M, float* __restrict__ d) {
  int id = blockIdx.x * blockDim.x + threadIdx.x;     // over 64*512
  if (id >= 64 * 512) return;
  int b = id >> 9, i = id & 511;
  const float* Mb = M + (long long)b * 65536;
  float rs = 1.f;                                     // identity diagonal
  if (i < 256) { for (int t = 0; t < 256; ++t) rs += Mb[i * 256 + t]; }
  else { int j = i - 256; for (int s = 0; s < 256; ++s) rs += Mb[s * 256 + j]; }
  d[id] = (rs > 0.f) ? rsqrtf(fmaxf(rs, 1e-12f)) : 0.f;
}

// y = d_j * xw : f32 master + bf16 GEMM operand
__global__ void scale_y(const float* __restrict__ xw, const float* __restrict__ d,
                        float* __restrict__ y, __bf16* __restrict__ y16) {
  long long n = 64ll * 512 * 128;
  for (long long i = (long long)blockIdx.x * blockDim.x + threadIdx.x; i < n;
       i += (long long)gridDim.x * blockDim.x) {
    float v = d[i >> 7] * xw[i];
    y[i] = v;
    y16[i] = f2bf(v);
  }
}

__global__ void gcn_epilogue(const float* __restrict__ gout,
                             const float* __restrict__ y,
                             const float* __restrict__ d,
                             float* __restrict__ g) {
  long long n = 64ll * 512 * 128;
  for (long long i = (long long)blockIdx.x * blockDim.x + threadIdx.x; i < n;
       i += (long long)gridDim.x * blockDim.x) {
    g[i] = fmaxf(0.f, d[i >> 7] * (gout[i] + y[i]));   // + identity-block term
  }
}

// MaxPool3x3 + AvgPool3x3 over [512,128], FC (42->3), row-max, softmax.
__global__ __launch_bounds__(192) void pool_fc_softmax(
    const float* __restrict__ g, const float* __restrict__ fc_w,
    const float* __restrict__ fc_b, float* __restrict__ out)
{
  int b = blockIdx.x, tid = threadIdx.x;
  __shared__ float red[3 * 192];
  float loc[3] = {-1e30f, -1e30f, -1e30f};
  if (tid < 170) {                              // Ho = 512/3 = 170
    float fcm[3], fca[3];
    for (int c = 0; c < 3; ++c) { fcm[c] = fc_b[c]; fca[c] = fc_b[c]; }
    const float* gb = g + (long long)b * 512 * 128;
    for (int wo = 0; wo < 42; ++wo) {           // Wo = 128/3 = 42
      float mx = -1e30f, sm = 0.f;
      for (int dy = 0; dy < 3; ++dy) {
        const float* row = gb + (3 * tid + dy) * 128 + 3 * wo;
        for (int dx = 0; dx < 3; ++dx) { float v = row[dx]; mx = fmaxf(mx, v); sm += v; }
      }
      float av = sm * (1.f / 9.f);
      for (int c = 0; c < 3; ++c) {
        fcm[c] += mx * fc_w[c * 42 + wo];
        fca[c] += av * fc_w[c * 42 + wo];
      }
    }
    for (int c = 0; c < 3; ++c) loc[c] = fmaxf(fcm[c], fca[c]);
  }
  for (int c = 0; c < 3; ++c) red[c * 192 + tid] = loc[c];
  __syncthreads();
  if (tid == 0) {
    float mx[3];
    for (int c = 0; c < 3; ++c) {
      float m = -1e30f;
      for (int i = 0; i < 192; ++i) m = fmaxf(m, red[c * 192 + i]);
      mx[c] = m;
    }
    float mm = fmaxf(mx[0], fmaxf(mx[1], mx[2]));
    float e0 = __expf(mx[0] - mm), e1 = __expf(mx[1] - mm), e2 = __expf(mx[2] - mm);
    float inv = 1.f / (e0 + e1 + e2);
    out[b * 3 + 0] = e0 * inv; out[b * 3 + 1] = e1 * inv; out[b * 3 + 2] = e2 * inv;
  }
}

// ---------------------------------------------------------------------------
// Host orchestration
// ---------------------------------------------------------------------------
extern "C" void kernel_launch(void* const* d_in, const int* in_sizes, int n_in,
                              void* d_out, int out_size, void* d_ws, size_t ws_size,
                              hipStream_t stream) {
  (void)in_sizes; (void)n_in; (void)out_size; (void)ws_size;
  const float* x1    = (const float*)d_in[0];
  const float* x2    = (const float*)d_in[1];
  const float* w_ih0 = (const float*)d_in[2];   // [2,512,300]
  const float* w_hh0 = (const float*)d_in[3];   // [2,512,128]
  const float* b_ih0 = (const float*)d_in[4];   // [2,512]
  const float* b_hh0 = (const float*)d_in[5];
  const float* w_ih  = (const float*)d_in[6];   // [2,2,512,256]
  const float* w_hh  = (const float*)d_in[7];   // [2,2,512,128]
  const float* b_ih  = (const float*)d_in[8];   // [2,2,512]
  const float* b_hh  = (const float*)d_in[9];
  const float* gcn_w = (const float*)d_in[10];  // [256,128]
  const float* fc_w  = (const float*)d_in[11];  // [3,42]
  const float* fc_b  = (const float*)d_in[12];  // [3]
  float* out = (float*)d_out;

  const int S = 256;
  const long long SB320 = 256ll * 64 * 320;
  const long long SB512 = 256ll * 64 * 512;
  const long long SB256 = 256ll * 64 * 256;

  char* base = (char*)d_ws;
  size_t off = 0;
  auto allocf = [&](long long n) {
    off = (off + 63) & ~(size_t)63; float* p = (float*)(base + off);
    off += (size_t)n * 4; return p;
  };
  auto allocb = [&](long long n) {
    off = (off + 63) & ~(size_t)63; __bf16* p = (__bf16*)(base + off);
    off += (size_t)n * 2; return p;
  };
  __bf16* xTb   = allocb(2 * SB320);        // [2][S][B][320] bf16
  __bf16* w0pb  = allocb(2ll * 512 * 320);  // padded w_ih0 bf16
  __bf16* wihb  = allocb(2ll * 2 * 512 * 256); // w_ih bf16
  float*  xpre  = allocf(4 * SB512);        // [2 inp][2 dir][S][B][512] f32
  float*  bufA  = allocf(2 * SB256);        // f32 layer out (ping)
  float*  bufB  = allocf(2 * SB256);        // f32 layer out (pong)
  __bf16* bufAb = allocb(2 * SB256);        // bf16 layer out (ping)
  __bf16* bufBb = allocb(2 * SB256);        // bf16 layer out (pong)
  float*  Mb    = allocf(64ll * 256 * 256); // cosine matrix f32
  __bf16* Mbb   = allocb(64ll * 256 * 256); // cosine matrix bf16
  __bf16* MbTb  = allocb(64ll * 256 * 256); // transposed cosine matrix bf16
  __bf16* gcnwTb= allocb(128 * 256);
  float*  inv1  = allocf(64 * 256);
  float*  inv2  = allocf(64 * 256);
  float*  dv    = allocf(64 * 512);
  float*  xw    = allocf(64ll * 512 * 128);
  float*  yb    = allocf(64ll * 512 * 128);
  __bf16* ybb   = allocb(64ll * 512 * 128);
  float*  gout  = allocf(64ll * 512 * 128);
  float*  gfin  = allocf(64ll * 512 * 128);

  auto gemm = [&](const __bf16* A, long long sA, int lda,
                  const __bf16* W, long long sW, int ldw,
                  const float* bb1, const float* bb2,
                  float* C, long long sC, int ldc,
                  int M, int N, int K, int batch) {
    dim3 grid(N / 128, M / 32, batch), blk(256);
    gemm_wmma<<<grid, blk, 0, stream>>>(A, sA, lda, W, sW, ldw,
                                        bb1, bb2, C, sC, ldc, K);
  };

  // ---- staging: transpose+pad inputs, pad/convert weights ----
  transpose_pad<<<4096, 256, 0, stream>>>(x1, xTb,         S, 64);
  transpose_pad<<<4096, 256, 0, stream>>>(x2, xTb + SB320, S, 64);
  pad_w0<<<640, 256, 0, stream>>>(w_ih0, w0pb);
  convert_bf<<<1024, 256, 0, stream>>>(w_ih, wihb, 2ll * 2 * 512 * 256);
  transpose_gcnw<<<128, 256, 0, stream>>>(gcn_w, gcnwTb);

  // ---- layer 0: xpre = x @ w_ih0^T + b_ih0 + b_hh0 (batched over inputs) ----
  for (int dir = 0; dir < 2; ++dir)
    gemm(xTb, SB320, 320,
         w0pb + (long long)dir * 512 * 320, 0, 320,
         b_ih0 + dir * 512, b_hh0 + dir * 512,
         xpre + (long long)dir * SB512, 2 * SB512, 512,
         16384, 512, 320, 2);
  lstm_layer<<<4, 512, 0, stream>>>(xpre, w_hh0, bufA, bufAb, S);

  // ---- layers 1,2 ----
  const __bf16* linb = bufAb; __bf16* loutb = bufBb;
  float* lout = bufB; float* lin = bufA;
  for (int l = 0; l < 2; ++l) {
    for (int dir = 0; dir < 2; ++dir)
      gemm(linb, SB256, 256,
           wihb + (long long)(l * 2 + dir) * 512 * 256, 0, 256,
           b_ih + (l * 2 + dir) * 512, b_hh + (l * 2 + dir) * 512,
           xpre + (long long)dir * SB512, 2 * SB512, 512,
           16384, 512, 256, 2);
    lstm_layer<<<4, 512, 0, stream>>>(xpre, w_hh + (long long)l * 2 * 512 * 128,
                                      lout, loutb, S);
    { const __bf16* tb = linb; linb = loutb; loutb = (__bf16*)tb; }
    { float* tf = lin; lin = lout; lout = tf; }
  }
  const float*  o1  = lin;                 // f32 [S][B][256]
  const float*  o2  = lin + SB256;
  const __bf16* o1b = linb;                // bf16 twins
  const __bf16* o2b = linb + SB256;

  // ---- cosine similarity M[b,s,t] ----
  row_invnorm<<<64, 256, 0, stream>>>(o1, inv1);
  row_invnorm<<<64, 256, 0, stream>>>(o2, inv2);
  gemm(o1b, 256, 16384, o2b, 256, 16384, nullptr, nullptr,
       Mb, 65536, 256, 256, 256, 256, 64);
  scale_M<<<4096, 256, 0, stream>>>(Mb, Mbb, inv1, inv2);
  degree_d<<<128, 256, 0, stream>>>(Mb, dv);
  transpose_M<<<4096, 256, 0, stream>>>(Mb, MbTb);

  // ---- xw = lstm_out @ gcn_w ----
  gemm(o1b, 256, 16384, gcnwTb, 0, 256, nullptr, nullptr,
       xw,             65536, 128, 256, 128, 256, 64);
  gemm(o2b, 256, 16384, gcnwTb, 0, 256, nullptr, nullptr,
       xw + 256 * 128, 65536, 128, 256, 128, 256, 64);

  // ---- y = d_j * xw;  A@y via block structure [[I,M],[M^T,I]] ----
  scale_y<<<4096, 256, 0, stream>>>(xw, dv, yb, ybb);
  gemm(Mbb,  65536, 256, ybb + 256 * 128, 65536, 128, nullptr, nullptr,
       gout,             65536, 128, 256, 128, 256, 64);   // top = M @ y_bot
  gemm(MbTb, 65536, 256, ybb,             65536, 128, nullptr, nullptr,
       gout + 256 * 128, 65536, 128, 256, 128, 256, 64);   // bot = M^T @ y_top
  gcn_epilogue<<<4096, 256, 0, stream>>>(gout, yb, dv, gfin);

  // ---- pooling + FC + row-max + softmax ----
  pool_fc_softmax<<<64, 192, 0, stream>>>(gfin, fc_w, fc_b, out);
}
